// Model_52518860096503
// MI455X (gfx1250) — matrix operand order repaired and checked by reading, not compiled, not run on hardware
//
#include <hip/hip_runtime.h>

// ---------------------------------------------------------------------------
// Problem constants (reference: B=8, S=4096, D=1024, scale = 1/sqrt(1024))
// ---------------------------------------------------------------------------
#define NBATCH 8
#define SEQ    4096
#define DIM    1024
#define NK     32          // keys per flash-attention tile (= WMMA K for PV)
#define BQ     32          // query rows per workgroup (2 row-groups x 16)
#define DQ     256         // D columns owned per wave (4 quarters)
#define SOFT_SCALE 0.03125f // 1/sqrt(1024)

typedef __attribute__((ext_vector_type(16))) __bf16    v16bf;
typedef __attribute__((ext_vector_type(8)))  float     v8f;
typedef __attribute__((ext_vector_type(4)))  float     fv4;
typedef __attribute__((ext_vector_type(4)))  unsigned  uv4;
typedef __attribute__((ext_vector_type(2)))  unsigned  uv2;

union Frag {                 // one 16x32 bf16 WMMA operand (8 VGPRs / lane)
    v16bf v;
    uv4   q[2];
    unsigned short s[16];
};

__device__ __forceinline__ unsigned short f2bf(float f) {
    unsigned u = __float_as_uint(f);
    u += 0x7FFFu + ((u >> 16) & 1u);           // round-to-nearest-even
    return (unsigned short)(u >> 16);
}

// A-matrix fragment (16x32 bf16) from a row-major bf16 LDS tile.
// Lane m (= lane&15) holds row m; VGPR0-3 = k in [hi*8, hi*8+8),
// VGPR4-7 = k in [16+hi*8, 16+hi*8+8), hi = lane>>4.  Two b128 loads / lane.
__device__ __forceinline__ Frag ld_frag_a(const unsigned short* base, int ld,
                                          int row0, int col0) {
    const int lane = threadIdx.x & 31;
    const int m = lane & 15, hi = lane >> 4;
    const unsigned short* p = base + (row0 + m) * ld + col0 + hi * 8;
    Frag f;
    f.q[0] = *(const uv4*)(p);
    f.q[1] = *(const uv4*)(p + 16);
    return f;
}

// B-matrix fragment (32x16 bf16), B[k,j] = base[(row0+j)*ld + col0 + k].
// Lane j holds column j; VGPR0-3 = k in [hi*8, hi*8+8), VGPR4-7 = k in [16+hi*8, 16+hi*8+8).
__device__ __forceinline__ Frag ld_frag_b(const unsigned short* base, int ld,
                                          int row0, int col0) {
    const int lane = threadIdx.x & 31;
    const int j = lane & 15, hi = lane >> 4;
    const unsigned short* p = base + (row0 + j) * ld + col0 + hi * 8;
    Frag f;
    f.q[0] = *(const uv4*)(p);
    f.q[1] = *(const uv4*)(p + 16);
    return f;
}

// A-fragment loaded straight from global fp32 (Q preload), converted to bf16.
__device__ __forceinline__ Frag ld_frag_a_g32(const float* base, int ld, int col0) {
    const int lane = threadIdx.x & 31;
    const int m = lane & 15, hi = lane >> 4;
    const float* p = base + (size_t)m * ld + col0 + hi * 8;
    Frag f;
#pragma unroll
    for (int g = 0; g < 2; ++g) {
        fv4 a = *(const fv4*)(p + g * 16);
        fv4 b = *(const fv4*)(p + g * 16 + 4);
        f.s[g*8+0] = f2bf(a.x); f.s[g*8+1] = f2bf(a.y);
        f.s[g*8+2] = f2bf(a.z); f.s[g*8+3] = f2bf(a.w);
        f.s[g*8+4] = f2bf(b.x); f.s[g*8+5] = f2bf(b.y);
        f.s[g*8+6] = f2bf(b.z); f.s[g*8+7] = f2bf(b.w);
    }
    return f;
}

// ---------------------------------------------------------------------------
// Kernel 1: flash attention.  Writes normalized context into d_out (fp32).
// Block = 256 threads = 8 waves arranged 2x4: wave (wrow, wd): wrow owns 16
// query rows, wd owns a 256-wide D quarter of the ctx accumulator.
// ~240 VGPRs/wave and 146 KB LDS -> 4 waves/SIMD, 2 workgroups/WGP.
// ---------------------------------------------------------------------------
__global__ void __launch_bounds__(256, 4)
attn_flash_kernel(const float* __restrict__ Qg, const float* __restrict__ Kg,
                  const float* __restrict__ Vg, float* __restrict__ Ctx)
{
    extern __shared__ unsigned char smem_raw[];
    unsigned short* Ksh = (unsigned short*)smem_raw;               // [NK][DIM] bf16
    unsigned short* Vt  = Ksh + NK * DIM;                          // [DIM][NK] bf16 (V^T)
    float*          SP  = (float*)(Vt + DIM * NK);                 // [8][16][NK] partial S
    unsigned short* Pb  = (unsigned short*)(SP + 8 * 16 * NK);     // [2][16][NK] P bf16

    const int tid  = threadIdx.x;
    const int lane = tid & 31;
    const int wave = tid >> 5;
    const int wrow = wave >> 2;      // 0..1: 16-row group
    const int wd   = wave & 3;       // 0..3: 256-wide D quarter
    const int n    = lane & 15;
    const int hi   = lane >> 4;

    const int b  = blockIdx.x / (SEQ / BQ);
    const int qt = blockIdx.x % (SEQ / BQ);

    const float* Qbase = Qg + ((size_t)b * SEQ + (size_t)qt * BQ + wrow * 16) * DIM;
    const float* Kb    = Kg + (size_t)b * SEQ * DIM;
    const float* Vb    = Vg + (size_t)b * SEQ * DIM;

    // Q fragments for this wave: 16 rows x 256 dims, bf16, kept in VGPRs (64).
    Frag qf[8];
#pragma unroll
    for (int f = 0; f < 8; ++f)
        qf[f] = ld_frag_a_g32(Qbase, DIM, wd * DQ + f * 32);

    v8f acc[16];                          // ctx accumulator: 16 x 256 f32
#pragma unroll
    for (int t = 0; t < 16; ++t)
#pragma unroll
        for (int v = 0; v < 8; ++v) acc[t][v] = 0.0f;

    float rm = -1e30f;                    // running max for row (lane&15)
    float rl = 0.0f;                      // running sum-of-exp

    for (int kt = 0; kt < SEQ / NK; ++kt) {
        const int key0 = kt * NK;

        // ---- stage K tile (row-major bf16) and V tile (transposed bf16) ----
        for (int it = tid; it < NK * (DIM / 4); it += 256) {
            int r  = it >> 8;                    // DIM/4 == 256
            int c4 = (it & 255) << 2;
            fv4 x = *(const fv4*)(Kb + (size_t)(key0 + r) * DIM + c4);
            uv2 pk;
            pk.x = (unsigned)f2bf(x.x) | ((unsigned)f2bf(x.y) << 16);
            pk.y = (unsigned)f2bf(x.z) | ((unsigned)f2bf(x.w) << 16);
            *(uv2*)(Ksh + r * DIM + c4) = pk;
        }
        for (int it = tid; it < NK * (DIM / 4); it += 256) {
            int r  = it >> 8;
            int c4 = (it & 255) << 2;
            fv4 x = *(const fv4*)(Vb + (size_t)(key0 + r) * DIM + c4);
            Vt[(c4 + 0) * NK + r] = f2bf(x.x);
            Vt[(c4 + 1) * NK + r] = f2bf(x.y);
            Vt[(c4 + 2) * NK + r] = f2bf(x.z);
            Vt[(c4 + 3) * NK + r] = f2bf(x.w);
        }
        // prefetch next tile into cache while we compute on this one
        if (kt + 1 < SEQ / NK) {
            size_t nb = (size_t)(key0 + NK) * DIM;
#pragma unroll
            for (int p = 0; p < 4; ++p) {
                __builtin_prefetch(Kb + nb + ((size_t)tid * 4 + p) * 32, 0, 0);
                __builtin_prefetch(Vb + nb + ((size_t)tid * 4 + p) * 32, 0, 0);
            }
        }
        __syncthreads();

        // ---- partial scores: S_q = Q(16xDQ) . K^T -> 16x32 f32 (one nt live) ----
        float* sp = SP + ((wrow * 4 + wd) * 16) * NK;
#pragma unroll
        for (int nt = 0; nt < 2; ++nt) {
            v8f sc;
#pragma unroll
            for (int v = 0; v < 8; ++v) sc[v] = 0.0f;
#pragma unroll
            for (int kk = 0; kk < 8; ++kk) {
                Frag bk = ld_frag_b(Ksh, DIM, nt * 16, wd * DQ + kk * 32);
                sc = __builtin_amdgcn_wmma_f32_16x16x32_bf16(
                    false, qf[kk].v, false, bk.v, (short)0, sc, false, false);
            }
#pragma unroll
            for (int v = 0; v < 8; ++v)
                sp[(v + 8 * hi) * NK + nt * 16 + n] = sc[v];
        }
        __syncthreads();

        // ---- online softmax, one row per lane (row = lane&15, duplicated) ----
        const float* p0 = SP + ((wrow * 4 + 0) * 16 + n) * NK;
        const float* p1 = SP + ((wrow * 4 + 1) * 16 + n) * NK;
        const float* p2 = SP + ((wrow * 4 + 2) * 16 + n) * NK;
        const float* p3 = SP + ((wrow * 4 + 3) * 16 + n) * NK;
        float mx = rm;
#pragma unroll
        for (int j = 0; j < NK; ++j)
            mx = fmaxf(mx, (p0[j] + p1[j] + p2[j] + p3[j]) * SOFT_SCALE);
        float rescale = __expf(rm - mx);
        float sum = 0.0f;
        unsigned short* pb = Pb + (wrow * 16 + n) * NK;
#pragma unroll
        for (int j = 0; j < NK; ++j) {
            float e = __expf((p0[j] + p1[j] + p2[j] + p3[j]) * SOFT_SCALE - mx);
            sum += e;
            if (wd == 0) pb[j] = f2bf(e);     // one wave per row-group writes P
        }
        const bool max_moved = (mx > rm);
        rl = rl * rescale + sum;
        rm = mx;
        __syncthreads();

        // ---- rescale accumulator (only when some row max moved) ----
        if (__any(max_moved)) {
#pragma unroll
            for (int v = 0; v < 8; ++v) {
                float fsc = __shfl(rescale, v + 8 * hi, 32);  // row (v+8*hi) factor
#pragma unroll
                for (int t = 0; t < 16; ++t) acc[t][v] *= fsc;
            }
        }
        // ---- ctx += P(16x32) @ V(32xDQ) ----
        Frag pa = ld_frag_a(Pb + wrow * 16 * NK, NK, 0, 0);
#pragma unroll
        for (int t = 0; t < 16; ++t) {
            Frag vb = ld_frag_b(Vt, NK, wd * DQ + t * 16, 0);  // B[k,j]=V[k][col]
            acc[t] = __builtin_amdgcn_wmma_f32_16x16x32_bf16(
                false, pa.v, false, vb.v, (short)0, acc[t], false, false);
        }
        __syncthreads();   // LDS reused next iteration
    }

    // ---- normalize rows and store context (fp32) into d_out ----
    float invl = 1.0f / rl;
    float* Cbase = Ctx + ((size_t)b * SEQ + (size_t)qt * BQ + wrow * 16) * DIM + wd * DQ;
#pragma unroll
    for (int v = 0; v < 8; ++v) {
        float fsc = __shfl(invl, v + 8 * hi, 32);
#pragma unroll
        for (int t = 0; t < 16; ++t)
            Cbase[(size_t)(v + 8 * hi) * DIM + t * 16 + n] = acc[t][v] * fsc;
    }
}

// ---------------------------------------------------------------------------
// Kernel 2: out = ctx @ W^T + b, in place on d_out.
// Block owns 32 rows exclusively; waves arranged 2x4 like kernel 1.
// All reads of those rows complete (barrier-ordered) before epilogue stores.
// ---------------------------------------------------------------------------
__global__ void __launch_bounds__(256, 4)
proj_kernel(float* __restrict__ Out, const float* __restrict__ Wg,
            const float* __restrict__ Bg)
{
    extern __shared__ unsigned char smem_raw[];
    unsigned short* Wsh = (unsigned short*)smem_raw;   // [DIM][32] bf16 (rows = e)
    unsigned short* Csh = Wsh + DIM * 32;              // [32][32]  bf16 (rows = i)

    const int tid  = threadIdx.x;
    const int lane = tid & 31;
    const int wave = tid >> 5;
    const int wrow = wave >> 2;      // 16-row group
    const int wd   = wave & 3;       // 256-wide output-column quarter
    const size_t row0 = (size_t)blockIdx.x * BQ;

    v8f acc[16];
#pragma unroll
    for (int t = 0; t < 16; ++t)
#pragma unroll
        for (int v = 0; v < 8; ++v) acc[t][v] = 0.0f;

    for (int kc = 0; kc < DIM / 32; ++kc) {
        const int d0 = kc * 32;
        // stage W chunk: all 1024 e-rows x 32 d (L2-resident, 4 MB total)
        for (int it = tid; it < DIM * 8; it += 256) {
            int e  = it >> 3;
            int c4 = (it & 7) << 2;
            fv4 x = *(const fv4*)(Wg + (size_t)e * DIM + d0 + c4);
            uv2 pk;
            pk.x = (unsigned)f2bf(x.x) | ((unsigned)f2bf(x.y) << 16);
            pk.y = (unsigned)f2bf(x.z) | ((unsigned)f2bf(x.w) << 16);
            *(uv2*)(Wsh + e * 32 + c4) = pk;
        }
        // stage ctx chunk: our 32 rows x 32 d (read-before-write, in place)
        for (int it = tid; it < BQ * 8; it += 256) {
            int r  = it >> 3;
            int c4 = (it & 7) << 2;
            fv4 x = *(const fv4*)(Out + (row0 + r) * DIM + d0 + c4);
            uv2 pk;
            pk.x = (unsigned)f2bf(x.x) | ((unsigned)f2bf(x.y) << 16);
            pk.y = (unsigned)f2bf(x.z) | ((unsigned)f2bf(x.w) << 16);
            *(uv2*)(Csh + r * 32 + c4) = pk;
        }
        __syncthreads();

        Frag a = ld_frag_a(Csh, 32, wrow * 16, 0);
#pragma unroll
        for (int t = 0; t < 16; ++t) {
            Frag bf = ld_frag_b(Wsh, 32, wd * DQ + t * 16, 0); // B[k,j]=W[e0+j][d0+k]
            acc[t] = __builtin_amdgcn_wmma_f32_16x16x32_bf16(false, a.v, false, bf.v,
                                                             (short)0, acc[t], false, false);
        }
        __syncthreads();
    }

    // epilogue: add bias, store
    const int n = lane & 15, hi = lane >> 4;
    float* Obase = Out + (row0 + wrow * 16) * DIM;
#pragma unroll
    for (int t = 0; t < 16; ++t) {
        int e = wd * DQ + t * 16 + n;
        float bias = Bg[e];
#pragma unroll
        for (int v = 0; v < 8; ++v)
            Obase[(size_t)(v + 8 * hi) * DIM + e] = acc[t][v] + bias;
    }
}

// ---------------------------------------------------------------------------
extern "C" void kernel_launch(void* const* d_in, const int* in_sizes, int n_in,
                              void* d_out, int out_size, void* d_ws, size_t ws_size,
                              hipStream_t stream) {
    (void)in_sizes; (void)n_in; (void)out_size; (void)d_ws; (void)ws_size;
    const float* Q = (const float*)d_in[0];
    const float* K = (const float*)d_in[1];
    const float* V = (const float*)d_in[2];
    const float* W = (const float*)d_in[3];
    const float* b = (const float*)d_in[4];
    float* Out = (float*)d_out;

    // LDS: K 64K + V^T 64K + score exchange 16K + P 2K = 146 KB  (2 WGs / WGP)
    size_t smem1 = (size_t)(NK * DIM) * 2 + (size_t)(DIM * NK) * 2
                 + (size_t)(8 * 16 * NK) * 4 + (size_t)(2 * 16 * NK) * 2;
    dim3 grid1(NBATCH * (SEQ / BQ));                  // 1024 blocks
    attn_flash_kernel<<<grid1, 256, smem1, stream>>>(Q, K, V, Out);

    size_t smem2 = (size_t)(DIM * 32) * 2 + (size_t)(BQ * 32) * 2;   // 66 KB
    dim3 grid2((NBATCH * SEQ) / BQ);                  // 1024 blocks
    proj_kernel<<<grid2, 256, smem2, stream>>>(Out, W, b);
}
